// MoE_14396730376783
// MI455X (gfx1250) — matrix operand order repaired and checked by reading, not compiled, not run on hardware
//
#include <hip/hip_runtime.h>
#include <hip/hip_bf16.h>

#define N_TOK 8192
#define D_IN  1024
#define D_OUT 1024
#define NE    16
#define EW_STRIDE (NE * D_OUT)   // 16384
#define NOISE_STD 0.1f
#define GRN_EPS 1e-6f
#define MTILE 64                 // rows per block (4 x 16 WMMA sub-tiles)

typedef __attribute__((ext_vector_type(16))) __bf16 v16bf;
typedef __attribute__((ext_vector_type(8)))  float  v8f;
typedef __attribute__((ext_vector_type(4)))  int    v4i;

#if defined(__HIP_DEVICE_COMPILE__) && defined(__has_builtin)
#  if __has_builtin(__builtin_amdgcn_global_load_async_to_lds_b128)
#    define HAVE_ASYNC_LDS 1
#  endif
#endif
#ifndef HAVE_ASYNC_LDS
#  define HAVE_ASYNC_LDS 0
#endif

#if HAVE_ASYNC_LDS
__device__ __forceinline__ void async_copy_b128(const void* g, void* l) {
    __builtin_amdgcn_global_load_async_to_lds_b128((v4i*)g, (v4i*)l, 0, 0);
}
__device__ __forceinline__ void wait_async0() {
#if __has_builtin(__builtin_amdgcn_s_wait_asynccnt)
    __builtin_amdgcn_s_wait_asynccnt(0);
#else
    asm volatile("s_wait_asynccnt 0x0" ::: "memory");
#endif
}
#endif

// ---------------------------------------------------------------------------
// Kernel 1: gate logits (x @ gate_w + b + 0.1*noise) + atomic sum of row norms
// ---------------------------------------------------------------------------
__global__ void __launch_bounds__(256)
moe_gate_logits(const float* __restrict__ x,
                const float* __restrict__ noise,
                const float* __restrict__ gate_w,
                const float* __restrict__ gate_b,
                float* __restrict__ logits,
                float* __restrict__ gx_sum) {
    const int lane = threadIdx.x & 31;
    const int wv   = threadIdx.x >> 5;
    const int t    = blockIdx.x * 8 + wv;
    const int e    = lane & 15;
    const int half = lane >> 4;

    const float* xr = x + (size_t)t * D_IN;
    const int k0 = half * (D_IN / 2);
    float acc = 0.f;
#pragma unroll 8
    for (int k = 0; k < D_IN / 2; ++k)
        acc = fmaf(xr[k0 + k], gate_w[(size_t)(k0 + k) * NE + e], acc);
    acc += __shfl_xor(acc, 16, 32);

    float g = acc + gate_b[e] + NOISE_STD * noise[(size_t)t * NE + e];
    if (lane < 16) logits[(size_t)t * NE + e] = g;

    float ss = (lane < 16) ? g * g : 0.f;
#pragma unroll
    for (int off = 16; off > 0; off >>= 1) ss += __shfl_xor(ss, off, 32);
    if (lane == 0) atomicAdd(gx_sum, sqrtf(ss));
}

// ---------------------------------------------------------------------------
// Kernel 2: GRN -> softmax -> top-2 -> scatter into expert buckets
// ---------------------------------------------------------------------------
__global__ void __launch_bounds__(256)
moe_gate_select(const float* __restrict__ logits,
                const float* __restrict__ gamma,
                const float* __restrict__ beta,
                const float* __restrict__ gx_sum,
                int*   __restrict__ cursor,
                int*   __restrict__ btok,
                float* __restrict__ bprob) {
    const int t = blockIdx.x * blockDim.x + threadIdx.x;
    if (t >= N_TOK) return;

    float g[NE];
    float ss = 0.f;
#pragma unroll
    for (int e = 0; e < NE; ++e) { g[e] = logits[(size_t)t * NE + e]; ss += g[e] * g[e]; }
    const float Gx   = sqrtf(ss);
    const float mean = gx_sum[0] * (1.0f / (float)N_TOK);
    const float Nx   = Gx / (mean + GRN_EPS);

    float mx = -3.0e38f;
#pragma unroll
    for (int e = 0; e < NE; ++e) { g[e] = gamma[e] * (g[e] * Nx) + beta[e]; mx = fmaxf(mx, g[e]); }
    float s = 0.f;
#pragma unroll
    for (int e = 0; e < NE; ++e) { g[e] = __expf(g[e] - mx); s += g[e]; }
    const float inv = 1.0f / s;
#pragma unroll
    for (int e = 0; e < NE; ++e) g[e] *= inv;

    int i0 = 0; float p0 = g[0];
#pragma unroll
    for (int e = 1; e < NE; ++e) if (g[e] > p0) { p0 = g[e]; i0 = e; }
    int i1 = (i0 == 0) ? 1 : 0; float p1 = g[i1];
#pragma unroll
    for (int e = 0; e < NE; ++e) if (e != i0 && g[e] > p1) { p1 = g[e]; i1 = e; }

    int pos0 = atomicAdd(&cursor[i0], 1);
    btok [i0 * N_TOK + pos0] = t;
    bprob[i0 * N_TOK + pos0] = p0;
    int pos1 = atomicAdd(&cursor[i1], 1);
    btok [i1 * N_TOK + pos1] = t;
    bprob[i1 * N_TOK + pos1] = p1;
}

// ---------------------------------------------------------------------------
// Kernel C1: x (f32) -> xbf (bf16), elementwise, 4/thread
// ---------------------------------------------------------------------------
__global__ void __launch_bounds__(256)
cvt_x_bf16(const float* __restrict__ in, __bf16* __restrict__ outb) {
    size_t i = ((size_t)blockIdx.x * 256 + threadIdx.x) * 4;
    float4 v = *reinterpret_cast<const float4*>(in + i);
    outb[i + 0] = (__bf16)v.x;
    outb[i + 1] = (__bf16)v.y;
    outb[i + 2] = (__bf16)v.z;
    outb[i + 3] = (__bf16)v.w;
}

// ---------------------------------------------------------------------------
// Kernel C2: expert_w [k][e*D_OUT+n] (f32) -> wbfT [e][n][k] (bf16), 32x32
// LDS-tiled transpose so both global read and write are coalesced.
// ---------------------------------------------------------------------------
__global__ void __launch_bounds__(256)
cvt_w_bf16_T(const float* __restrict__ w, __bf16* __restrict__ wT) {
    __shared__ float tile[32][33];
    const int e     = blockIdx.z;
    const int kbase = blockIdx.x * 32;
    const int nbase = blockIdx.y * 32;
    const int tid   = threadIdx.x;
#pragma unroll
    for (int i = 0; i < 4; ++i) {
        int flat = tid + 256 * i;
        int kr = flat >> 5, nc = flat & 31;
        tile[kr][nc] = w[(size_t)(kbase + kr) * EW_STRIDE + e * D_OUT + nbase + nc];
    }
    __syncthreads();
#pragma unroll
    for (int i = 0; i < 4; ++i) {
        int flat = tid + 256 * i;
        int nr = flat >> 5, kc = flat & 31;
        wT[((size_t)e * D_OUT + nbase + nr) * D_IN + kbase + kc] = (__bf16)tile[kc][nr];
    }
}

// ---------------------------------------------------------------------------
// Kernel 3 (fast path): grouped GEMM, 64x128 tile, pre-converted bf16 operands.
// B tiles staged with global_load_async_to_lds_b128 (ASYNCcnt) when available.
// ---------------------------------------------------------------------------
__global__ void __launch_bounds__(256)
moe_expert_gemm_bf16(const __bf16* __restrict__ xbf,
                     const __bf16* __restrict__ wbfT,
                     const float*  __restrict__ expert_b,
                     const int*    __restrict__ cursor,
                     const int*    __restrict__ btok,
                     const float*  __restrict__ bprob,
                     float* __restrict__ out) {
    const int e     = blockIdx.z;
    const int mt    = blockIdx.x;
    const int nbase = blockIdx.y * 128;
    const int cnt   = cursor[e];
    if (mt * MTILE >= cnt) return;

    __shared__ __align__(32) __bf16 ldsA[4 * 16 * 32];   // 4 sub-tiles, lane-permuted
    __shared__ __align__(32) __bf16 ldsB[128 * 32];      // [n][k], contiguous rows
    __shared__ int   tokS[MTILE];
    __shared__ float prbS[MTILE];

    const int tid = threadIdx.x;
    if (tid < MTILE) {
        int r = mt * MTILE + tid;
        if (r < cnt) { tokS[tid] = btok[e * N_TOK + r]; prbS[tid] = bprob[e * N_TOK + r]; }
        else         { tokS[tid] = -1;                  prbS[tid] = 0.f; }
    }
    __syncthreads();

    const int wv       = tid >> 5;
    const int lane     = tid & 31;
    const int laneHalf = lane >> 4;
    const int laneMod  = lane & 15;

    // A staging coords: 2 consecutive k of row am per sub-tile (loads kept
    // unconditional: invalid tokens clamp to row 0, value zeroed by select).
    const int am    = tid >> 4;
    const int ak    = (tid & 15) * 2;
    const int aLane = am + (ak >> 4) * 16;
    const int aPos  = ak & 15;
    const __bf16* aPtr[4];
    bool aVal[4];
#pragma unroll
    for (int s = 0; s < 4; ++s) {
        int tok = tokS[s * 16 + am];
        aVal[s] = tok >= 0;
        aPtr[s] = xbf + (size_t)(aVal[s] ? tok : 0) * D_IN + ak;
    }

    // B async staging coords: each thread copies 32 contiguous bytes of one row.
    const int brow = tid >> 1;               // 0..127 (column n)
    const int bsub = (tid & 1) * 16;         // element offset within row
    const __bf16* bSrcRow = wbfT + (size_t)(nbase + brow) * D_IN + bsub;
    __bf16*       bDstRow = &ldsB[brow * 32 + bsub];

    const v16bf* pA = reinterpret_cast<const v16bf*>(ldsA);
    const v16bf* pB = reinterpret_cast<const v16bf*>(ldsB);
    const int bFrag = (wv * 16 + laneMod) * 2 + laneHalf;

    v8f acc[4];
#pragma unroll
    for (int s = 0; s < 4; ++s) acc[s] = (v8f){0.f,0.f,0.f,0.f,0.f,0.f,0.f,0.f};

    for (int k0 = 0; k0 < D_IN; k0 += 32) {
        // B tile: 128 rows x 32 bf16 -> LDS, async DMA (no VGPR round-trip).
#if HAVE_ASYNC_LDS
        async_copy_b128(bSrcRow + k0,     bDstRow);
        async_copy_b128(bSrcRow + k0 + 8, bDstRow + 8);
#else
        {
            uint4 v0 = *reinterpret_cast<const uint4*>(bSrcRow + k0);
            uint4 v1 = *reinterpret_cast<const uint4*>(bSrcRow + k0 + 8);
            *reinterpret_cast<uint4*>(bDstRow)     = v0;
            *reinterpret_cast<uint4*>(bDstRow + 8) = v1;
        }
#endif
        // A tile: 64 gathered rows x 32 k, already bf16; 2 elems (u32) per thread
        // per sub-tile, stored to the WMMA lane-permuted layout.
#pragma unroll
        for (int s = 0; s < 4; ++s) {
            unsigned int v = *reinterpret_cast<const unsigned int*>(aPtr[s] + k0);
            v = aVal[s] ? v : 0u;
            *reinterpret_cast<unsigned int*>(&ldsA[s * 512 + aLane * 16 + aPos]) = v;
        }
        if (k0 + 32 < D_IN) __builtin_prefetch(bSrcRow + k0 + 32, 0, 0);
#if HAVE_ASYNC_LDS
        wait_async0();
#endif
        __syncthreads();

        v16bf b = pB[bFrag];
#pragma unroll
        for (int s = 0; s < 4; ++s) {
            v16bf a = pA[s * 32 + lane];
            acc[s] = __builtin_amdgcn_wmma_f32_16x16x32_bf16(false, a, false, b,
                                                             (short)0, acc[s], false, false);
        }
        __syncthreads();
    }

    const int   colG = nbase + wv * 16 + laneMod;
    const float bias = expert_b[e * D_OUT + colG];
#pragma unroll
    for (int s = 0; s < 4; ++s) {
#pragma unroll
        for (int v = 0; v < 8; ++v) {
            int M   = s * 16 + v + laneHalf * 8;
            int tok = tokS[M];
            if (tok >= 0) {
                float val = prbS[M] * (acc[s][v] + bias);
                atomicAdd(&out[(size_t)tok * D_OUT + colG], val);
            }
        }
    }
}

// ---------------------------------------------------------------------------
// Kernel 3 (fallback, small workspace): f32 operands, convert in-loop.
// ---------------------------------------------------------------------------
__global__ void __launch_bounds__(256)
moe_expert_gemm(const float* __restrict__ x,
                const float* __restrict__ expert_w,
                const float* __restrict__ expert_b,
                const int*   __restrict__ cursor,
                const int*   __restrict__ btok,
                const float* __restrict__ bprob,
                float* __restrict__ out) {
    const int e     = blockIdx.z;
    const int mt    = blockIdx.x;
    const int nbase = blockIdx.y * 128;
    const int cnt   = cursor[e];
    if (mt * MTILE >= cnt) return;

    __shared__ __align__(32) __bf16 ldsA[4 * 16 * 32];
    __shared__ __align__(32) __bf16 ldsB[128 * 32];
    __shared__ int   tokS[MTILE];
    __shared__ float prbS[MTILE];

    const int tid = threadIdx.x;
    if (tid < MTILE) {
        int r = mt * MTILE + tid;
        if (r < cnt) { tokS[tid] = btok[e * N_TOK + r]; prbS[tid] = bprob[e * N_TOK + r]; }
        else         { tokS[tid] = -1;                  prbS[tid] = 0.f; }
    }
    __syncthreads();

    const int wv       = tid >> 5;
    const int lane     = tid & 31;
    const int laneHalf = lane >> 4;
    const int laneMod  = lane & 15;

    const int am    = tid >> 4;
    const int ak    = (tid & 15) * 2;
    const int aLane = am + (ak >> 4) * 16;
    const int aPos  = ak & 15;
    const float* aPtr[4];
    bool aVal[4];
#pragma unroll
    for (int s = 0; s < 4; ++s) {
        int tok = tokS[s * 16 + am];
        aVal[s] = tok >= 0;
        aPtr[s] = x + (size_t)(aVal[s] ? tok : 0) * D_IN + ak;
    }

    const v16bf* pA = reinterpret_cast<const v16bf*>(ldsA);
    const v16bf* pB = reinterpret_cast<const v16bf*>(ldsB);
    const int bFrag = (wv * 16 + laneMod) * 2 + laneHalf;

    v8f acc[4];
#pragma unroll
    for (int s = 0; s < 4; ++s) acc[s] = (v8f){0.f,0.f,0.f,0.f,0.f,0.f,0.f,0.f};

    const float* wBase = expert_w + (size_t)e * D_OUT + nbase;

    for (int k0 = 0; k0 < D_IN; k0 += 32) {
#pragma unroll
        for (int s = 0; s < 4; ++s) {
            float2 xv = *reinterpret_cast<const float2*>(aPtr[s] + k0);
            if (!aVal[s]) xv = make_float2(0.f, 0.f);
            __bf16* dst = &ldsA[s * 512 + aLane * 16 + aPos];
            dst[0] = (__bf16)xv.x;
            dst[1] = (__bf16)xv.y;
        }
#pragma unroll
        for (int i = 0; i < 16; ++i) {
            int flat = tid + 256 * i;
            int bk = flat >> 7;
            int bn = flat & 127;
            float w = wBase[(size_t)(k0 + bk) * EW_STRIDE + bn];
            ldsB[bn * 32 + bk] = (__bf16)w;
        }
        if (k0 + 32 < D_IN)
            __builtin_prefetch(wBase + (size_t)(k0 + 32 + (tid >> 3)) * EW_STRIDE
                               + ((tid & 7) << 4), 0, 0);
        __syncthreads();

        v16bf b = pB[bFrag];
#pragma unroll
        for (int s = 0; s < 4; ++s) {
            v16bf a = pA[s * 32 + lane];
            acc[s] = __builtin_amdgcn_wmma_f32_16x16x32_bf16(false, a, false, b,
                                                             (short)0, acc[s], false, false);
        }
        __syncthreads();
    }

    const int   colG = nbase + wv * 16 + laneMod;
    const float bias = expert_b[e * D_OUT + colG];
#pragma unroll
    for (int s = 0; s < 4; ++s) {
#pragma unroll
        for (int v = 0; v < 8; ++v) {
            int M   = s * 16 + v + laneHalf * 8;
            int tok = tokS[M];
            if (tok >= 0) {
                float val = prbS[M] * (acc[s][v] + bias);
                atomicAdd(&out[(size_t)tok * D_OUT + colG], val);
            }
        }
    }
}

// ---------------------------------------------------------------------------
extern "C" void kernel_launch(void* const* d_in, const int* in_sizes, int n_in,
                              void* d_out, int out_size, void* d_ws, size_t ws_size,
                              hipStream_t stream) {
    const float* x        = (const float*)d_in[0];
    const float* noise    = (const float*)d_in[1];
    const float* gate_w   = (const float*)d_in[2];
    const float* gate_b   = (const float*)d_in[3];
    const float* expert_w = (const float*)d_in[4];
    const float* expert_b = (const float*)d_in[5];
    const float* gamma    = (const float*)d_in[6];
    const float* beta     = (const float*)d_in[7];
    float* out = (float*)d_out;

    // Workspace layout: [0]=gx_sum, [1..16]=cursors, pad to 32 dwords;
    // logits [N*E]; btok [N*E]; bprob [N*E]; then (fast path) xbf, wbfT.
    float* ws     = (float*)d_ws;
    float* gx_sum = ws;
    int*   cursor = (int*)(ws + 1);
    float* logits = ws + 32;
    int*   btok   = (int*)  (ws + 32 + (size_t)N_TOK * NE);
    float* bprob  =          ws + 32 + (size_t)2 * N_TOK * NE;

    const size_t smallBytes = (32 + (size_t)3 * N_TOK * NE) * sizeof(float);
    const size_t xbfBytes   = (size_t)N_TOK * D_IN * sizeof(__bf16);
    const size_t wbfBytes   = (size_t)NE * D_OUT * D_IN * sizeof(__bf16);
    const bool   fast       = ws_size >= smallBytes + xbfBytes + wbfBytes;
    __bf16* xbf  = (__bf16*)((char*)d_ws + smallBytes);
    __bf16* wbfT = (__bf16*)((char*)d_ws + smallBytes + xbfBytes);

    (void)hipMemsetAsync(d_ws, 0, 32 * sizeof(float), stream);
    (void)hipMemsetAsync(d_out, 0, (size_t)N_TOK * D_OUT * sizeof(float), stream);

    moe_gate_logits<<<N_TOK / 8, 256, 0, stream>>>(x, noise, gate_w, gate_b, logits, gx_sum);
    moe_gate_select<<<N_TOK / 256, 256, 0, stream>>>(logits, gamma, beta, gx_sum,
                                                     cursor, btok, bprob);
    dim3 grid(N_TOK / MTILE, D_OUT / 128, NE);
    if (fast) {
        cvt_x_bf16<<<(N_TOK * D_IN) / (256 * 4), 256, 0, stream>>>(x, xbf);
        dim3 gw(D_IN / 32, D_OUT / 32, NE);
        cvt_w_bf16_T<<<gw, 256, 0, stream>>>(expert_w, wbfT);
        moe_expert_gemm_bf16<<<grid, 256, 0, stream>>>(xbf, wbfT, expert_b,
                                                       cursor, btok, bprob, out);
    } else {
        moe_expert_gemm<<<grid, 256, 0, stream>>>(x, expert_w, expert_b,
                                                  cursor, btok, bprob, out);
    }
}
